// LSTM_32959579030294
// MI455X (gfx1250) — matrix operand order repaired and checked by reading, not compiled, not run on hardware
//
#include <hip/hip_runtime.h>
#include <hip/hip_bf16.h>

typedef __attribute__((ext_vector_type(16))) _Float16 v16h;
typedef __attribute__((ext_vector_type(8)))  float    v8f;

#define Tn 128
#define Bn 256
#define En 128
#define Hn 1024
#define Cn 10
#define G4 (4*Hn)

// hidden-A fragment buffer: 16 m-tiles x 32 k-tiles, 512 halfs (1024B) per frag
#define HA_HALFS (16*32*512)

// ---------------------------------------------------------------------------
// Zero-init state + both hidden-A ping-pong buffers (ws is poisoned 0xAA).
// ---------------------------------------------------------------------------
__global__ void zero_ws_kernel(float* __restrict__ state,
                               unsigned* __restrict__ hA0,
                               unsigned* __restrict__ hA1) {
    int i = blockIdx.x * blockDim.x + threadIdx.x;      // 0 .. 262143
    if (i < Bn * Hn) state[i] = 0.0f;
    if (i < HA_HALFS / 2) { hA0[i] = 0u; hA1[i] = 0u; } // halfs -> dwords
}

// ---------------------------------------------------------------------------
// Pack Wh (4 gates, each [H,H] f32 row-major, W[k*H+n]) into WMMA B-fragment
// layout, f16.  Fragment order: [ht(64)][kt(32)][g(4)] so one wave streams a
// contiguous 128KB region over its K loop (4 gate frags adjacent per kt).
// B 32x16 layout: lane l -> n = l%16, h2 = l/16;
// VGPR v, pack p -> k = 16*(v>>2) + 8*h2 + 2*(v&3) + p  (within the 32-wide K tile).
// One thread per (frag, lane) writes 16 halfs (32 contiguous bytes).
// ---------------------------------------------------------------------------
__global__ void pack_wh_kernel(_Float16* __restrict__ WhB,
                               const float* __restrict__ Wgh,
                               const float* __restrict__ Wih,
                               const float* __restrict__ Wfh,
                               const float* __restrict__ Woh) {
    int tid  = blockIdx.x * blockDim.x + threadIdx.x;   // 0 .. 262143
    int lane = tid & 31;
    int frag = tid >> 5;                                // (ht*32 + kt)*4 + g
    int g  = frag & 3;
    int kt = (frag >> 2) & 31;
    int ht = frag >> 7;
    const float* W = (g == 0) ? Wgh : (g == 1) ? Wih : (g == 2) ? Wfh : Woh;
    int n  = ht * 16 + (lane & 15);
    int h2 = lane >> 4;
    _Float16* dst = WhB + (size_t)frag * 512 + lane * 16;
#pragma unroll
    for (int v = 0; v < 8; ++v) {
#pragma unroll
        for (int p = 0; p < 2; ++p) {
            int k = kt * 32 + 16 * (v >> 2) + h2 * 8 + 2 * (v & 3) + p;
            dst[v * 2 + p] = (_Float16)W[k * Hn + n];
        }
    }
}

// ---------------------------------------------------------------------------
// P[v][j] = sum_e emb[v,e] * Wx_gate[e, j%H] + b_gate[j%H]   (10 x 4096)
// Eliminates the per-step input GEMM (vocab is only 10).
// ---------------------------------------------------------------------------
__global__ void compute_p_kernel(float* __restrict__ P,
                                 const float* __restrict__ emb,
                                 const float* __restrict__ Wgx,
                                 const float* __restrict__ Wix,
                                 const float* __restrict__ Wfx,
                                 const float* __restrict__ Wox,
                                 const float* __restrict__ bg,
                                 const float* __restrict__ bi,
                                 const float* __restrict__ bf,
                                 const float* __restrict__ bo) {
    int tid = blockIdx.x * blockDim.x + threadIdx.x;    // 0 .. 40959
    if (tid >= 10 * G4) return;
    int j = tid % G4;
    int vv = tid / G4;
    int g  = j >> 10;
    int jg = j & 1023;
    const float* W  = (g == 0) ? Wgx : (g == 1) ? Wix : (g == 2) ? Wfx : Wox;
    const float* bb = (g == 0) ? bg  : (g == 1) ? bi  : (g == 2) ? bf  : bo;
    float s = bb[jg];
#pragma unroll 4
    for (int e = 0; e < En; ++e) s += emb[vv * En + e] * W[e * Hn + jg];
    P[tid] = s;
}

// ---------------------------------------------------------------------------
// One LSTM timestep.  512 waves (64 blocks x 8 waves).
// Block stages TWO A panels (64KB, one mt-pair shared by all 8 waves) in LDS;
// wave (mt-pair, ht) streams its 128KB B region once and does 8 WMMAs per
// 4KB of B (2 m-tiles x 4 gates) -> half the L2 bandwidth per unit of math.
// Nonlinear state update entirely in registers; new hidden repacked into the
// next step's A-fragment layout (ping-pong).
// A 16x32 layout: lane l -> m = l%16, h = l/16;
//   VGPR v, pack p -> k = 16*(v>>2) + 8*h + 2*(v&3) + p.
// ---------------------------------------------------------------------------
__global__ void __launch_bounds__(256)
lstm_step_kernel(const _Float16* __restrict__ WhB,
                 const float* __restrict__ P,
                 const int* __restrict__ xt,          // x[t, :], 256 ints
                 float* __restrict__ state,           // [256,1024] in-place
                 float* __restrict__ hidden,          // [256,1024] f32 out
                 const _Float16* __restrict__ hA_in,  // packed A frags (prev h)
                 _Float16* __restrict__ hA_out) {     // packed A frags (new h)
    __shared__ _Float16 Alds[32 * 1024];              // 64 KB: 2 panels

    int tid  = threadIdx.x;
    int lane = tid & 31;
    int wv   = tid >> 5;
    int mtp  = blockIdx.x & 7;                  // m-tile pair 0..7
    int mt0  = mtp * 2;                         // even m-tile of the pair
    int ht   = (blockIdx.x >> 3) * 8 + wv;      // hidden tile  0..63

    // Cooperative stage of both A panels: 64KB, 256B (16 x uint4) per thread.
    {
        const uint4* src = (const uint4*)((const char*)hA_in +
                                          (size_t)mt0 * 32768) + tid * 16;
        uint4* dst = (uint4*)Alds + tid * 16;
#pragma unroll
        for (int q = 0; q < 16; ++q) dst[q] = src[q];
    }
    __syncthreads();

    v8f acc[2][4];
#pragma unroll
    for (int m2 = 0; m2 < 2; ++m2)
#pragma unroll
        for (int g = 0; g < 4; ++g)
            acc[m2][g] = (v8f){0.f,0.f,0.f,0.f,0.f,0.f,0.f,0.f};

    const char* Bb = (const char*)WhB + (size_t)ht * 131072 + (size_t)lane * 32;
    const v16h* Av0 = (const v16h*)Alds;              // panel mt0: 32 frags
    const v16h* Av1 = (const v16h*)Alds + 32 * 32;    // panel mt0+1

#pragma unroll
    for (int kt = 0; kt < 32; ++kt) {
        const char* p = Bb + kt * 4096;
        v16h b0 = *(const v16h*)(p);
        v16h b1 = *(const v16h*)(p + 1024);
        v16h b2 = *(const v16h*)(p + 2048);
        v16h b3 = *(const v16h*)(p + 3072);
        if (kt + 3 < 32)                        // warm near cache ahead
            __builtin_prefetch((const void*)(Bb + (kt + 3) * 4096), 0, 3);

        v16h a0 = Av0[kt * 32 + lane];          // ds_load from LDS
        v16h a1 = Av1[kt * 32 + lane];

        acc[0][0] = __builtin_amdgcn_wmma_f32_16x16x32_f16(false, a0, false, b0,
                                              (short)0, acc[0][0], false, false);
        acc[0][1] = __builtin_amdgcn_wmma_f32_16x16x32_f16(false, a0, false, b1,
                                              (short)0, acc[0][1], false, false);
        acc[0][2] = __builtin_amdgcn_wmma_f32_16x16x32_f16(false, a0, false, b2,
                                              (short)0, acc[0][2], false, false);
        acc[0][3] = __builtin_amdgcn_wmma_f32_16x16x32_f16(false, a0, false, b3,
                                              (short)0, acc[0][3], false, false);
        acc[1][0] = __builtin_amdgcn_wmma_f32_16x16x32_f16(false, a1, false, b0,
                                              (short)0, acc[1][0], false, false);
        acc[1][1] = __builtin_amdgcn_wmma_f32_16x16x32_f16(false, a1, false, b1,
                                              (short)0, acc[1][1], false, false);
        acc[1][2] = __builtin_amdgcn_wmma_f32_16x16x32_f16(false, a1, false, b2,
                                              (short)0, acc[1][2], false, false);
        acc[1][3] = __builtin_amdgcn_wmma_f32_16x16x32_f16(false, a1, false, b3,
                                              (short)0, acc[1][3], false, false);
    }

    // C/D layout: VGPR r, lane l -> row m = r + 8*(l/16), col n = l%16.
    int half = lane >> 4;
    int n    = lane & 15;
    int hc   = ht * 16 + n;                     // hidden unit 0..1023
    // A-fragment inverse mapping for repacking the new hidden value at k = hc
    int kt2 = hc >> 5;
    int kk  = hc & 31;
    int h2  = (kk >> 3) & 1;
    int v   = ((kk >> 4) << 2) + ((kk & 7) >> 1);
    int p   = kk & 1;

#pragma unroll
    for (int m2 = 0; m2 < 2; ++m2) {
        int mt = mt0 + m2;
#pragma unroll
        for (int r = 0; r < 8; ++r) {
            int row = mt * 16 + 8 * half + r;   // batch index
            int idx = xt[row];
            const float* Pr = P + (size_t)idx * G4 + hc;
            float pg = acc[m2][0][r] + Pr[0 * Hn];
            float pi = acc[m2][1][r] + Pr[1 * Hn];
            float pf = acc[m2][2][r] + Pr[2 * Hn];
            float po = acc[m2][3][r] + Pr[3 * Hn];
            float gg = tanhf(pg);
            float ii = 1.0f / (1.0f + __expf(-pi));
            float ff = 1.0f / (1.0f + __expf(-pf));
            float oo = 1.0f / (1.0f + __expf(-po));
            float s  = state[row * Hn + hc];
            float sn = gg * ii + s * ff;
            state[row * Hn + hc] = sn;
            float hd = tanhf(sn) * oo;
            hidden[row * Hn + hc] = hd;
            int lane2 = (row & 15) + (h2 << 4);
            size_t off = (size_t)((row >> 4) * 32 + kt2) * 512
                       + lane2 * 16 + v * 2 + p;
            hA_out[off] = (_Float16)hd;
        }
    }
}

// ---------------------------------------------------------------------------
// Final projection: out[b,c] = hidden[b,:] @ Wph[:,c] + bp[c]   (2560 outputs)
// ---------------------------------------------------------------------------
__global__ void proj_kernel(float* __restrict__ out,
                            const float* __restrict__ hidden,
                            const float* __restrict__ Wph,
                            const float* __restrict__ bp) {
    int tid = blockIdx.x * blockDim.x + threadIdx.x;
    if (tid >= Bn * Cn) return;
    int b = tid / Cn, c = tid % Cn;
    float s = bp[c];
#pragma unroll 4
    for (int h = 0; h < Hn; ++h) s += hidden[b * Hn + h] * Wph[h * Cn + c];
    out[tid] = s;
}

// ---------------------------------------------------------------------------
extern "C" void kernel_launch(void* const* d_in, const int* in_sizes, int n_in,
                              void* d_out, int out_size, void* d_ws, size_t ws_size,
                              hipStream_t stream) {
    const int*   x   = (const int*)  d_in[0];
    const float* emb = (const float*)d_in[1];
    const float* Wgx = (const float*)d_in[2];
    const float* Wgh = (const float*)d_in[3];
    const float* Wix = (const float*)d_in[4];
    const float* Wih = (const float*)d_in[5];
    const float* Wfx = (const float*)d_in[6];
    const float* Wfh = (const float*)d_in[7];
    const float* Wox = (const float*)d_in[8];
    const float* Woh = (const float*)d_in[9];
    const float* bg  = (const float*)d_in[10];
    const float* bi  = (const float*)d_in[11];
    const float* bf  = (const float*)d_in[12];
    const float* bo  = (const float*)d_in[13];
    const float* Wph = (const float*)d_in[14];
    const float* bp  = (const float*)d_in[15];

    char* ws = (char*)d_ws;
    _Float16* WhB    = (_Float16*)ws;                              // 8 MB
    float*    P      = (float*)(ws + (8u << 20));                  // 160 KB
    float*    state  = (float*)(ws + (8u << 20) + (256u << 10));   // 1 MB
    float*    hidden = state + Bn * Hn;                            // 1 MB
    _Float16* hA0    = (_Float16*)(hidden + Bn * Hn);              // 512 KB
    _Float16* hA1    = hA0 + HA_HALFS;                             // 512 KB

    zero_ws_kernel  <<<1024, 256, 0, stream>>>(state, (unsigned*)hA0, (unsigned*)hA1);
    pack_wh_kernel  <<<1024, 256, 0, stream>>>(WhB, Wgh, Wih, Wfh, Woh);
    compute_p_kernel<<<160,  256, 0, stream>>>(P, emb, Wgx, Wix, Wfx, Wox,
                                               bg, bi, bf, bo);
    for (int t = 0; t < Tn; ++t) {
        const _Float16* in  = (t & 1) ? hA1 : hA0;
        _Float16*       out = (t & 1) ? hA0 : hA1;
        lstm_step_kernel<<<64, 256, 0, stream>>>(WhB, P, x + t * Bn,
                                                 state, hidden, in, out);
    }
    proj_kernel<<<10, 256, 0, stream>>>((float*)d_out, hidden, Wph, bp);
}
